// Decoder_23210003267889
// MI455X (gfx1250) — compile-verified, matrix operand order checked by reading, not run on hardware
//
#include <hip/hip_runtime.h>
#include <hip/hip_bf16.h>

// ---------------------------------------------------------------------------
// Problem constants (from reference)
// ---------------------------------------------------------------------------
#define NPTS 65536
#define LAT  256
#define CCH  512
#define FF   128
#define HID  512
#define DIN  768          // CCH + 2*FF
#define KAUG 2048         // 8 corners * LAT
#define EPS  1e-5f

// Try the CDNA5 async Global->LDS DMA path (ASYNCcnt-tracked, no VGPR staging).
// If the assembler rejects the mnemonic, flip to 0 (plain load + ds_store).
#define USE_ASYNC_COPY 1

typedef __attribute__((ext_vector_type(16))) __bf16 v16bf;
typedef __attribute__((ext_vector_type(8)))  float  v8f;

union FragBF {            // 32 bytes = 16 bf16 = one WMMA A/B fragment per lane
    uint4  u[2];
    v16bf  v;
};

__device__ __forceinline__ void copy16_g2lds(void* lds_ptr, const void* gptr) {
#if USE_ASYNC_COPY
    unsigned lds_addr = (unsigned)(unsigned long long)lds_ptr; // low 32 bits of
    // a generic LDS address are the LDS byte address (ISA 10.2 aperture rules)
    asm volatile("global_load_async_to_lds_b128 %0, %1, off"
                 :: "v"(lds_addr), "v"(gptr) : "memory");
#else
    *(uint4*)lds_ptr = *(const uint4*)gptr;
#endif
}

__device__ __forceinline__ void async_copy_fence() {
#if USE_ASYNC_COPY
    asm volatile("s_wait_asynccnt 0" ::: "memory");
#endif
}

// ---------------------------------------------------------------------------
// Workspace layout (bytes)
// ---------------------------------------------------------------------------
#define OFF_W8   ((size_t)0)                          // N*8 f32            = 2 MB
#define OFF_X0   (OFF_W8  + (size_t)NPTS*8*4)         // N*768 bf16         = 96 MB
#define OFF_X1   (OFF_X0  + (size_t)NPTS*DIN*2)       // N*512 bf16         = 64 MB
#define OFF_WALL (OFF_X1  + (size_t)NPTS*HID*2)       // 512*2048 bf16      = 2 MB
#define OFF_W0T  (OFF_WALL+ (size_t)CCH*KAUG*2)       // 512*768 bf16
#define OFF_WHT  (OFF_W0T + (size_t)HID*DIN*2)        // 7*512*512 bf16

// ---------------------------------------------------------------------------
// K1: convert + transpose all GEMM weights to bf16, N-major (B-frag friendly)
// ---------------------------------------------------------------------------
#define N_W0T  (HID*DIN)          // 393216
#define N_WHT  (7*HID*HID)        // 1835008
#define N_WALL (CCH*KAUG)         // 1048576
#define N_PREP (N_W0T + N_WHT + N_WALL)

__global__ __launch_bounds__(256) void prep_weights(
    const float* __restrict__ W0, const float* __restrict__ Wh,
    const float* __restrict__ convT_w,
    __bf16* __restrict__ W0t, __bf16* __restrict__ Wht, __bf16* __restrict__ Wallt)
{
    int idx = blockIdx.x * 256 + threadIdx.x;
    if (idx < N_W0T) {
        int n = idx / DIN, k = idx % DIN;
        W0t[idx] = (__bf16)W0[k * HID + n];
    } else if (idx < N_W0T + N_WHT) {
        int j = idx - N_W0T;
        int l = j / (HID * HID);
        int r = j - l * (HID * HID);
        int n = r / HID, k = r % HID;
        Wht[j] = (__bf16)Wh[l * HID * HID + k * HID + n];
    } else if (idx < N_PREP) {
        int j = idx - (N_W0T + N_WHT);
        int c = j >> 11;                 // /2048
        int rest = j & 2047;
        int corner = rest >> 8;
        int i = rest & 255;
        Wallt[j] = (__bf16)convT_w[((size_t)(i * CCH + c)) * 8 + corner];
    }
}

// ---------------------------------------------------------------------------
// K2: per-point trilinear corner weights + Fourier features (x0 cols 512..767)
// ---------------------------------------------------------------------------
__global__ __launch_bounds__(128) void point_prep(
    const float* __restrict__ input, const float* __restrict__ gauss,
    float* __restrict__ w8, __bf16* __restrict__ x0)
{
    int b = blockIdx.x;
    int t = threadIdx.x;
    const float* row = input + (size_t)b * (LAT + 3);
    float px = row[LAT + 0], py = row[LAT + 1], pz = row[LAT + 2];

    if (t < FF) {
        const float* g = gauss + t * 3;
        float ang = 6.28318530717958647f * (px * g[0] + py * g[1] + pz * g[2]);
        float s, c;
        __sincosf(ang, &s, &c);
        __bf16* xr = x0 + (size_t)b * DIN;
        xr[CCH + t]      = (__bf16)s;
        xr[CCH + FF + t] = (__bf16)c;
    }
    if (t < 8) {
        float fx = (px + 1.f) * 0.5f;
        float fy = (py + 1.f) * 0.5f;
        float fz = (pz + 1.f) * 0.5f;
        float wx = (t & 4) ? fx : 1.f - fx;
        float wy = (t & 2) ? fy : 1.f - fy;
        float wz = (t & 1) ? fz : 1.f - fz;
        w8[(size_t)b * 8 + t] = wx * wy * wz;
    }
}

// ---------------------------------------------------------------------------
// K3: corner GEMM.  lat_i[m,c] = sum_{cr,k} (w8[m,cr]*lat[m,k]) * Wall[c,cr,k]
// Augmented A (32 x 2048) built in LDS.  8 waves, each: 32 rows x 64 cols
// = 2(M) x 4(N) acc tiles -> per K-step: 4 ds + 8 global loads for 8 WMMAs.
// ---------------------------------------------------------------------------
#define AUG_LDA 2056   // 2048 + 8 bf16 pad -> +4 banks/row, conflict-free b128

__global__ __launch_bounds__(256) void corner_gemm(
    const float* __restrict__ input, const float* __restrict__ w8,
    const __bf16* __restrict__ Wallt, const float* __restrict__ convT_b,
    __bf16* __restrict__ Y /* x0, ld = DIN */)
{
    extern __shared__ __bf16 As[];   // 32 * AUG_LDA

    const int tid  = threadIdx.x;
    const int lane = tid & 31;
    const int wave = tid >> 5;       // 0..7  -> N eighth (64 cols)
    const int ln15 = lane & 15;
    const int hi   = lane >> 4;
    const int row0b = blockIdx.x * 32;

    // Build augmented A tile in LDS: As[m][cr*256+k] = input[m][k] * w8[m][cr]
    for (int idx = tid; idx < 32 * KAUG; idx += 256) {
        int m  = idx >> 11;
        int kk = idx & 2047;
        int corner = kk >> 8;
        int k      = kk & 255;
        float v = input[(size_t)(row0b + m) * (LAT + 3) + k]
                * w8[(size_t)(row0b + m) * 8 + corner];
        As[m * AUG_LDA + kk] = (__bf16)v;
    }
    __syncthreads();

    v8f acc[2][4];
    for (int mt = 0; mt < 2; ++mt)
        for (int nt = 0; nt < 4; ++nt)
            for (int v = 0; v < 8; ++v) acc[mt][nt][v] = 0.f;

    const int n0 = wave * 64;
    const __bf16* arow0 = As + (ln15)      * AUG_LDA + 8 * hi;
    const __bf16* arow1 = As + (16 + ln15) * AUG_LDA + 8 * hi;

    for (int k0 = 0; k0 < KAUG; k0 += 32) {
        FragBF a0, a1;
        a0.u[0] = *(const uint4*)(arow0 + k0);
        a0.u[1] = *(const uint4*)(arow0 + k0 + 16);
        a1.u[0] = *(const uint4*)(arow1 + k0);
        a1.u[1] = *(const uint4*)(arow1 + k0 + 16);
#pragma unroll
        for (int nt = 0; nt < 4; ++nt) {
            int n = n0 + nt * 16 + ln15;
            const __bf16* bcol = Wallt + (size_t)n * KAUG + k0 + 8 * hi;
            FragBF b;
            b.u[0] = *(const uint4*)(bcol);
            b.u[1] = *(const uint4*)(bcol + 16);
            acc[0][nt] = __builtin_amdgcn_wmma_f32_16x16x32_bf16(
                false, a0.v, false, b.v, (short)0, acc[0][nt], false, false);
            acc[1][nt] = __builtin_amdgcn_wmma_f32_16x16x32_bf16(
                false, a1.v, false, b.v, (short)0, acc[1][nt], false, false);
        }
    }

    // epilogue: + convT bias (corner weights sum to 1)
#pragma unroll
    for (int nt = 0; nt < 4; ++nt) {
        int n = n0 + nt * 16 + ln15;
        float bb = convT_b[n];
#pragma unroll
        for (int mt = 0; mt < 2; ++mt)
#pragma unroll
            for (int v = 0; v < 8; ++v) {
                int m = mt * 16 + v + 8 * hi;
                Y[(size_t)(row0b + m) * DIN + n] = (__bf16)(acc[mt][nt][v] + bb);
            }
    }
}

// ---------------------------------------------------------------------------
// K4: fused GEMM + bias + LayerNorm + ReLU layer (templated on K).
// Block tile 64 x 512; 8 waves: 2(M half of 32 rows) x 4(N quarter of 128).
// Wave = 2 M-tiles x 8 N-tiles = 16 acc tiles (128 VGPRs).
// A tile (64 x K) staged in LDS via async Global->LDS copy.
// ---------------------------------------------------------------------------
template <int K>
__global__ __launch_bounds__(256) void layer_kernel(
    const __bf16* __restrict__ X, int ldx,
    const __bf16* __restrict__ Wt,        // [HID][K] n-major
    const float* __restrict__ bias,
    const float* __restrict__ gamma, const float* __restrict__ beta,
    __bf16* __restrict__ Y, int ldy)
{
    constexpr int LDA = K + 8;            // +16B/row -> +4 banks, conflict-free
    extern __shared__ __bf16 As[];        // 64 * LDA
    __shared__ float red[4][64][2];       // [wn][row][sum,sumsq]

    const int tid  = threadIdx.x;
    const int lane = tid & 31;
    const int wave = tid >> 5;
    const int wm   = wave >> 2;           // 0..1  (32-row half)
    const int wn   = wave & 3;            // 0..3  (128-col quarter)
    const int ln15 = lane & 15;
    const int hi   = lane >> 4;

    const int row0b = blockIdx.x * 64;
    const int n0    = wn * 128;

    // ---- stage A tile (64 rows x K) into LDS, 16B chunks, coalesced ----
    constexpr int CPR    = K / 8;         // b128 chunks per row
    constexpr int CHUNKS = 64 * CPR;
    for (int c = tid; c < CHUNKS; c += 256) {
        int r = c / CPR;
        int j = c - r * CPR;
        copy16_g2lds(As + r * LDA + j * 8,
                     X + (size_t)(row0b + r) * ldx + j * 8);
    }
    async_copy_fence();
    __syncthreads();

    v8f acc[2][8];
    for (int mt = 0; mt < 2; ++mt)
        for (int nt = 0; nt < 8; ++nt)
            for (int v = 0; v < 8; ++v) acc[mt][nt][v] = 0.f;

    const __bf16* arow0 = As + (wm * 32 + ln15)      * LDA + 8 * hi;
    const __bf16* arow1 = As + (wm * 32 + 16 + ln15) * LDA + 8 * hi;

    for (int k0 = 0; k0 < K; k0 += 32) {
        FragBF a0, a1;
        a0.u[0] = *(const uint4*)(arow0 + k0);
        a0.u[1] = *(const uint4*)(arow0 + k0 + 16);
        a1.u[0] = *(const uint4*)(arow1 + k0);
        a1.u[1] = *(const uint4*)(arow1 + k0 + 16);
#pragma unroll
        for (int nt = 0; nt < 8; ++nt) {
            int n = n0 + nt * 16 + ln15;
            const __bf16* bcol = Wt + (size_t)n * K + k0 + 8 * hi;
            FragBF b;
            b.u[0] = *(const uint4*)(bcol);
            b.u[1] = *(const uint4*)(bcol + 16);
            acc[0][nt] = __builtin_amdgcn_wmma_f32_16x16x32_bf16(
                false, a0.v, false, b.v, (short)0, acc[0][nt], false, false);
            acc[1][nt] = __builtin_amdgcn_wmma_f32_16x16x32_bf16(
                false, a1.v, false, b.v, (short)0, acc[1][nt], false, false);
        }
    }

    // ---- bias + per-row partial LN stats ----
    float ps[2][8], pq[2][8];
    for (int mt = 0; mt < 2; ++mt)
        for (int v = 0; v < 8; ++v) { ps[mt][v] = 0.f; pq[mt][v] = 0.f; }
#pragma unroll
    for (int nt = 0; nt < 8; ++nt) {
        int n = n0 + nt * 16 + ln15;
        float bb = bias[n];
#pragma unroll
        for (int mt = 0; mt < 2; ++mt)
#pragma unroll
            for (int v = 0; v < 8; ++v) {
                float x = acc[mt][nt][v] + bb;
                acc[mt][nt][v] = x;
                ps[mt][v] += x;
                pq[mt][v] += x * x;
            }
    }
    // reduce across the 16 lanes of each half (xor masks stay within half)
#pragma unroll
    for (int m = 1; m <= 8; m <<= 1)
#pragma unroll
        for (int mt = 0; mt < 2; ++mt)
#pragma unroll
            for (int v = 0; v < 8; ++v) {
                ps[mt][v] += __shfl_xor(ps[mt][v], m, 32);
                pq[mt][v] += __shfl_xor(pq[mt][v], m, 32);
            }
    if (ln15 == 0) {
#pragma unroll
        for (int mt = 0; mt < 2; ++mt)
#pragma unroll
            for (int v = 0; v < 8; ++v) {
                int row = wm * 32 + mt * 16 + v + 8 * hi;
                red[wn][row][0] = ps[mt][v];
                red[wn][row][1] = pq[mt][v];
            }
    }
    __syncthreads();

    float mu[2][8], rs[2][8];
#pragma unroll
    for (int mt = 0; mt < 2; ++mt)
#pragma unroll
        for (int v = 0; v < 8; ++v) {
            int row = wm * 32 + mt * 16 + v + 8 * hi;
            float s = red[0][row][0] + red[1][row][0]
                    + red[2][row][0] + red[3][row][0];
            float q = red[0][row][1] + red[1][row][1]
                    + red[2][row][1] + red[3][row][1];
            float m_  = s * (1.f / HID);
            float var = q * (1.f / HID) - m_ * m_;
            mu[mt][v] = m_;
            rs[mt][v] = rsqrtf(var + EPS);
        }

    // ---- normalize + affine + relu + bf16 store ----
#pragma unroll
    for (int nt = 0; nt < 8; ++nt) {
        int n = n0 + nt * 16 + ln15;
        float g = gamma[n], be = beta[n];
#pragma unroll
        for (int mt = 0; mt < 2; ++mt)
#pragma unroll
            for (int v = 0; v < 8; ++v) {
                int row = wm * 32 + mt * 16 + v + 8 * hi;
                float y = (acc[mt][nt][v] - mu[mt][v]) * rs[mt][v] * g + be;
                y = fmaxf(y, 0.f);
                Y[(size_t)(row0b + row) * ldy + n] = (__bf16)y;
            }
    }
}

// ---------------------------------------------------------------------------
// K5: 512 -> 4 head.  One wave per point: lane covers 16 contiguous elems
// (2 coalesced b128 loads), shfl_xor reduce, lane 0 writes tanh / *255.
// ---------------------------------------------------------------------------
__global__ __launch_bounds__(256) void out_kernel(
    const __bf16* __restrict__ X, int ldx,
    const float* __restrict__ Wout, const float* __restrict__ bout,
    float* __restrict__ out)
{
    int wave = threadIdx.x >> 5;
    int lane = threadIdx.x & 31;
    int i = blockIdx.x * 8 + wave;

    const __bf16* x = X + (size_t)i * ldx + lane * 16;
    FragBF xf;
    xf.u[0] = *(const uint4*)(x);
    xf.u[1] = *(const uint4*)(x + 8);

    float a0 = 0.f, a1 = 0.f, a2 = 0.f, a3 = 0.f;
#pragma unroll
    for (int e = 0; e < 16; ++e) {
        float xv = (float)xf.v[e];
        const float* w = Wout + (lane * 16 + e) * 4;
        a0 += xv * w[0];
        a1 += xv * w[1];
        a2 += xv * w[2];
        a3 += xv * w[3];
    }
#pragma unroll
    for (int m = 1; m <= 16; m <<= 1) {
        a0 += __shfl_xor(a0, m, 32);
        a1 += __shfl_xor(a1, m, 32);
        a2 += __shfl_xor(a2, m, 32);
        a3 += __shfl_xor(a3, m, 32);
    }
    if (lane == 0) {
        float* o = out + (size_t)i * 4;
        o[0] = tanhf(a0 + bout[0]);
        o[1] = (a1 + bout[1]) * 255.f;
        o[2] = (a2 + bout[2]) * 255.f;
        o[3] = (a3 + bout[3]) * 255.f;
    }
}

// ---------------------------------------------------------------------------
// Host launcher
// ---------------------------------------------------------------------------
extern "C" void kernel_launch(void* const* d_in, const int* in_sizes, int n_in,
                              void* d_out, int out_size, void* d_ws, size_t ws_size,
                              hipStream_t stream) {
    const float* input   = (const float*)d_in[0];
    const float* convT_w = (const float*)d_in[1];
    const float* convT_b = (const float*)d_in[2];
    const float* gauss   = (const float*)d_in[3];
    const float* W0      = (const float*)d_in[4];
    const float* b0      = (const float*)d_in[5];
    const float* Wh      = (const float*)d_in[6];
    const float* bh      = (const float*)d_in[7];
    const float* ln_g    = (const float*)d_in[8];
    const float* ln_b    = (const float*)d_in[9];
    const float* W_out   = (const float*)d_in[10];
    const float* b_out   = (const float*)d_in[11];

    char* ws = (char*)d_ws;
    float*  w8    = (float*) (ws + OFF_W8);
    __bf16* x0    = (__bf16*)(ws + OFF_X0);   // N x 768
    __bf16* x1    = (__bf16*)(ws + OFF_X1);   // N x 512
    __bf16* Wallt = (__bf16*)(ws + OFF_WALL);
    __bf16* W0t   = (__bf16*)(ws + OFF_W0T);
    __bf16* Wht   = (__bf16*)(ws + OFF_WHT);

    prep_weights<<<(N_PREP + 255) / 256, 256, 0, stream>>>(W0, Wh, convT_w,
                                                           W0t, Wht, Wallt);
    point_prep<<<NPTS, 128, 0, stream>>>(input, gauss, w8, x0);

    corner_gemm<<<NPTS / 32, 256, 32 * AUG_LDA * sizeof(__bf16), stream>>>(
        input, w8, Wallt, convT_b, x0);

    // layer 0: x0(768) -> x1(512)
    layer_kernel<DIN><<<NPTS / 64, 256, 64 * (DIN + 8) * sizeof(__bf16), stream>>>(
        x0, DIN, W0t, b0, ln_g, ln_b, x1, HID);

    // hidden layers 0..6, ping-pong x1 <-> x0 (x0 row stride stays DIN)
    const __bf16* src = x1; int src_ld = HID;
    __bf16* dst = x0;       int dst_ld = DIN;
    for (int l = 0; l < 7; ++l) {
        layer_kernel<HID><<<NPTS / 64, 256, 64 * (HID + 8) * sizeof(__bf16), stream>>>(
            src, src_ld,
            Wht + (size_t)l * HID * HID,
            bh + (size_t)l * HID,
            ln_g + (size_t)(l + 1) * HID,
            ln_b + (size_t)(l + 1) * HID,
            dst, dst_ld);
        const __bf16* ts = src; int tl = src_ld;
        src = dst;  src_ld = dst_ld;
        dst = (__bf16*)ts; dst_ld = tl;
    }
    // final activations are in x0 (ld = DIN)

    out_kernel<<<NPTS / 8, 256, 0, stream>>>(x0, DIN, W_out, b_out,
                                             (float*)d_out);
}
